// ViTLayer_52226802320100
// MI455X (gfx1250) — compile-verified, matrix operand order checked by reading, not run on hardware
//
#include <hip/hip_runtime.h>

// Complex transformer layer for MI455X (gfx1250, wave32, WMMA).
// B=16 S=512 D=1024 H=16 DH=64 M=4096.
#define B_    16
#define S_    512
#define D_    1024
#define H_    16
#define M_    4096
#define ROWS_ (B_ * S_)   // 8192
#define D2_   (2 * D_)    // 2048 halves per interleaved complex row
#define EPS_  1e-5f

typedef _Float16 f16;
typedef _Float16 v16h __attribute__((ext_vector_type(16)));
typedef float    v8f  __attribute__((ext_vector_type(8)));

__device__ __forceinline__ v8f wmma32(v16h a, v16h b, v8f c) {
  // D = A(16x32 f16) * B(32x16 f16) + C(16x16 f32)
  return __builtin_amdgcn_wmma_f32_16x16x32_f16(false, a, false, b, (short)0, c,
                                                false, false);
}

// Fragment load matching the 16-bit A-matrix 16x32 VGPR layout:
// element e holds K = (e<8?0:16) + half*8 + (e%8) of one LDS row.
__device__ __forceinline__ v16h ldfrag(const f16* rowbase, int half) {
  union { v16h v; uint4 u[2]; } f;
  f.u[0] = *(const uint4*)(rowbase + half * 8);
  f.u[1] = *(const uint4*)(rowbase + 16 + half * 8);
  return f.v;
}

// ---------------------------------------------------------------------------
// Complex GEMM: C[M,N] = A[M,K]c * B[K,N]c + bias (+ residual), WMMA f16.
// A16: f16 interleaved [M, 2K] halves.  Bc: f32 complex [K,N].
// Tile: 128(M) x 64(N) per 256-thread block, K'-step 32 halves (16 complex).
// ---------------------------------------------------------------------------
template <bool OUT_F16, bool HAS_RES>
__global__ __launch_bounds__(256) void cgemm_wmma(
    const f16* __restrict__ A16, const float2* __restrict__ Bc,
    const float2* __restrict__ bias, const float2* __restrict__ res,
    void* __restrict__ outp, int K, int N) {
  __shared__ f16 sA[128][40];
  __shared__ f16 sB[2][64][40];  // [0]=[Br;-Bi] for Cr, [1]=[Bi;Br] for Ci
  const int tid = threadIdx.x;
  const int lane = tid & 31, half = (tid >> 4) & 1, l16 = tid & 15;
  const int wave = tid >> 5;
  const int wm = (wave & 3) * 32, wn = (wave >> 2) * 32;
  const int m0 = blockIdx.y * 128, n0 = blockIdx.x * 64;
  const size_t lda = (size_t)2 * K;
  (void)lane;
  v8f accR[2][2] = {};
  v8f accI[2][2] = {};
  for (int kc = 0; kc < K; kc += 16) {
    const int kk0 = kc * 2;
    // Stage A: 128 rows x 32 halves (contiguous 16B chunks).
#pragma unroll
    for (int i = 0; i < 2; ++i) {
      const int id = tid + i * 256;
      const int r = id >> 2, ch = (id & 3) << 3;
      *(uint4*)&sA[r][ch] =
          *(const uint4*)(A16 + (size_t)(m0 + r) * lda + (kk0 + ch));
    }
    if (kc + 16 < K)  // hint next A tile -> global_prefetch_b8
      __builtin_prefetch(
          A16 + (size_t)(m0 + (tid >> 1)) * lda + (kk0 + 32) + ((tid & 1) << 4),
          0, 0);
    // Stage B: 16 complex rows x 64 cols, transposed + sign-expanded.
#pragma unroll
    for (int i = 0; i < 4; ++i) {
      const int id = tid + i * 256;
      const int kk = id >> 6, n = id & 63;
      const float2 w = Bc[(size_t)(kc + kk) * N + (n0 + n)];
      sB[0][n][2 * kk]     = (f16)w.x;
      sB[0][n][2 * kk + 1] = (f16)(-w.y);
      sB[1][n][2 * kk]     = (f16)w.y;
      sB[1][n][2 * kk + 1] = (f16)w.x;
    }
    __syncthreads();
    const v16h af0 = ldfrag(&sA[wm + l16][0], half);
    const v16h af1 = ldfrag(&sA[wm + 16 + l16][0], half);
    const v16h br0 = ldfrag(&sB[0][wn + l16][0], half);
    const v16h br1 = ldfrag(&sB[0][wn + 16 + l16][0], half);
    const v16h bi0 = ldfrag(&sB[1][wn + l16][0], half);
    const v16h bi1 = ldfrag(&sB[1][wn + 16 + l16][0], half);
    accR[0][0] = wmma32(af0, br0, accR[0][0]);
    accR[0][1] = wmma32(af0, br1, accR[0][1]);
    accR[1][0] = wmma32(af1, br0, accR[1][0]);
    accR[1][1] = wmma32(af1, br1, accR[1][1]);
    accI[0][0] = wmma32(af0, bi0, accI[0][0]);
    accI[0][1] = wmma32(af0, bi1, accI[0][1]);
    accI[1][0] = wmma32(af1, bi0, accI[1][0]);
    accI[1][1] = wmma32(af1, bi1, accI[1][1]);
    __syncthreads();
  }
#pragma unroll
  for (int mf = 0; mf < 2; ++mf)
#pragma unroll
    for (int nf = 0; nf < 2; ++nf) {
      const int col = n0 + wn + nf * 16 + l16;
      const float2 bb = bias[col];
#pragma unroll
      for (int v = 0; v < 8; ++v) {
        const int row = m0 + wm + mf * 16 + half * 8 + v;
        float re = accR[mf][nf][v] + bb.x;
        float im = accI[mf][nf][v] + bb.y;
        if (HAS_RES) {
          const float2 rr = res[(size_t)row * N + col];
          re += rr.x;
          im += rr.y;
        }
        if (OUT_F16) {
          union { f16 h[2]; unsigned u; } p;
          p.h[0] = (f16)re;
          p.h[1] = (f16)im;
          *(unsigned*)((f16*)outp + ((size_t)row * N + col) * 2) = p.u;
        } else {
          ((float2*)outp)[(size_t)row * N + col] = make_float2(re, im);
        }
      }
    }
}

// ---------------------------------------------------------------------------
// Attention scores: s[bh,s,t] = dot(q128[s], k128[t]) * 1/sqrt(DH).
// Interleaved (re,im) dot == Re(q . conj(k)).  Real GEMM 512x512x128.
// ---------------------------------------------------------------------------
__global__ __launch_bounds__(256) void qk_scores(const f16* __restrict__ q16,
                                                 const f16* __restrict__ k16,
                                                 f16* __restrict__ s16,
                                                 float scale) {
  __shared__ f16 sA[128][40];
  __shared__ f16 sB[64][40];
  const int tid = threadIdx.x;
  const int half = (tid >> 4) & 1, l16 = tid & 15;
  const int wave = tid >> 5;
  const int wm = (wave & 3) * 32, wn = (wave >> 2) * 32;
  const int s0 = blockIdx.y * 128, t0 = blockIdx.x * 64;
  const int bh = blockIdx.z, b = bh >> 4, h = bh & 15;
  const f16* qb = q16 + (size_t)b * S_ * D2_ + h * 128;
  const f16* kb = k16 + (size_t)b * S_ * D2_ + h * 128;
  v8f acc[2][2] = {};
  for (int k0 = 0; k0 < 128; k0 += 32) {
#pragma unroll
    for (int i = 0; i < 2; ++i) {
      const int id = tid + i * 256;
      const int r = id >> 2, ch = (id & 3) << 3;
      *(uint4*)&sA[r][ch] = *(const uint4*)(qb + (size_t)(s0 + r) * D2_ + k0 + ch);
    }
    {
      const int n = tid >> 2, ch = (tid & 3) << 3;
      *(uint4*)&sB[n][ch] = *(const uint4*)(kb + (size_t)(t0 + n) * D2_ + k0 + ch);
    }
    __syncthreads();
    const v16h af0 = ldfrag(&sA[wm + l16][0], half);
    const v16h af1 = ldfrag(&sA[wm + 16 + l16][0], half);
    const v16h bf0 = ldfrag(&sB[wn + l16][0], half);
    const v16h bf1 = ldfrag(&sB[wn + 16 + l16][0], half);
    acc[0][0] = wmma32(af0, bf0, acc[0][0]);
    acc[0][1] = wmma32(af0, bf1, acc[0][1]);
    acc[1][0] = wmma32(af1, bf0, acc[1][0]);
    acc[1][1] = wmma32(af1, bf1, acc[1][1]);
    __syncthreads();
  }
  f16* srow = s16 + (size_t)bh * S_ * S_;
#pragma unroll
  for (int mf = 0; mf < 2; ++mf)
#pragma unroll
    for (int nf = 0; nf < 2; ++nf)
#pragma unroll
      for (int v = 0; v < 8; ++v)
        srow[(size_t)(s0 + wm + mf * 16 + half * 8 + v) * S_ +
             (t0 + wn + nf * 16 + l16)] = (f16)(acc[mf][nf][v] * scale);
}

// ---------------------------------------------------------------------------
// A*V: out[s,c] = sum_t probs[s,t] * v128[t,c]; c spans interleaved (vr,vi).
// Real GEMM 512 x 128 x 512 per (b,h).
// ---------------------------------------------------------------------------
__global__ __launch_bounds__(256) void av_wmma(const f16* __restrict__ p16,
                                               const f16* __restrict__ v16p,
                                               f16* __restrict__ ao16) {
  __shared__ f16 sA[128][40];
  __shared__ f16 sB[64][40];
  const int tid = threadIdx.x;
  const int half = (tid >> 4) & 1, l16 = tid & 15;
  const int wave = tid >> 5;
  const int wm = (wave & 3) * 32, wn = (wave >> 2) * 32;
  const int s0 = blockIdx.y * 128, n0 = blockIdx.x * 64;
  const int bh = blockIdx.z, b = bh >> 4, h = bh & 15;
  const f16* pa = p16 + (size_t)bh * S_ * S_;
  const f16* vb = v16p + (size_t)b * S_ * D2_ + h * 128;
  v8f acc[2][2] = {};
  for (int k0 = 0; k0 < S_; k0 += 32) {
#pragma unroll
    for (int i = 0; i < 2; ++i) {
      const int id = tid + i * 256;
      const int r = id >> 2, ch = (id & 3) << 3;
      *(uint4*)&sA[r][ch] = *(const uint4*)(pa + (size_t)(s0 + r) * S_ + k0 + ch);
    }
    {  // transpose-stage V tile: sB[n][k]
      const int kk = tid >> 3, nch = (tid & 7) << 3;
      union { uint4 u; f16 h[8]; } t;
      t.u = *(const uint4*)(vb + (size_t)(k0 + kk) * D2_ + n0 + nch);
#pragma unroll
      for (int j = 0; j < 8; ++j) sB[nch + j][kk] = t.h[j];
    }
    __syncthreads();
    const v16h af0 = ldfrag(&sA[wm + l16][0], half);
    const v16h af1 = ldfrag(&sA[wm + 16 + l16][0], half);
    const v16h bf0 = ldfrag(&sB[wn + l16][0], half);
    const v16h bf1 = ldfrag(&sB[wn + 16 + l16][0], half);
    acc[0][0] = wmma32(af0, bf0, acc[0][0]);
    acc[0][1] = wmma32(af0, bf1, acc[0][1]);
    acc[1][0] = wmma32(af1, bf0, acc[1][0]);
    acc[1][1] = wmma32(af1, bf1, acc[1][1]);
    __syncthreads();
  }
#pragma unroll
  for (int mf = 0; mf < 2; ++mf)
#pragma unroll
    for (int nf = 0; nf < 2; ++nf)
#pragma unroll
      for (int v = 0; v < 8; ++v) {
        const int row = s0 + wm + mf * 16 + half * 8 + v;
        const int col = n0 + wn + nf * 16 + l16;
        ao16[((size_t)b * S_ + row) * D2_ + h * 128 + col] = (f16)acc[mf][nf][v];
      }
}

// ---------------------------------------------------------------------------
// Complex LayerNorm (2x2 covariance whitening + affine + bias) -> f16 out.
// One block per row; works in-place when IN_F16 and in==out.
// ---------------------------------------------------------------------------
template <bool IN_F16>
__global__ __launch_bounds__(256) void ln_cplx(const void* __restrict__ inp,
                                               const float* __restrict__ w,
                                               const float2* __restrict__ bias,
                                               f16* __restrict__ out, int C) {
  const int row = blockIdx.x, tid = threadIdx.x;
  const float2* inf = (const float2*)inp + (size_t)row * C;
  const f16* inh = (const f16*)inp + (size_t)row * C * 2;
  float sr = 0, si = 0, srr = 0, sii = 0, sri = 0;
  for (int c = tid; c < C; c += 256) {
    float xr, xi;
    if (IN_F16) { xr = (float)inh[2 * c]; xi = (float)inh[2 * c + 1]; }
    else { const float2 t = inf[c]; xr = t.x; xi = t.y; }
    sr += xr; si += xi; srr += xr * xr; sii += xi * xi; sri += xr * xi;
  }
  __shared__ float red[5][256];
  red[0][tid] = sr; red[1][tid] = si; red[2][tid] = srr;
  red[3][tid] = sii; red[4][tid] = sri;
  __syncthreads();
  for (int s = 128; s > 0; s >>= 1) {
    if (tid < s) {
#pragma unroll
      for (int j = 0; j < 5; ++j) red[j][tid] += red[j][tid + s];
    }
    __syncthreads();
  }
  const float invC = 1.0f / (float)C;
  const float mur = red[0][0] * invC, mui = red[1][0] * invC;
  const float vrr = red[2][0] * invC - mur * mur + EPS_;
  const float vii = red[3][0] * invC - mui * mui + EPS_;
  const float vri = red[4][0] * invC - mur * mui;
  const float sd = sqrtf(vrr * vii - vri * vri);
  const float tt = sqrtf(vrr + vii + 2.0f * sd);
  const float inv = 1.0f / (sd * tt);
  const float wrr = (vii + sd) * inv, wii = (vrr + sd) * inv, wri = -vri * inv;
  f16* orow = out + (size_t)row * C * 2;
  for (int c = tid; c < C; c += 256) {
    float xr, xi;
    if (IN_F16) { xr = (float)inh[2 * c]; xi = (float)inh[2 * c + 1]; }
    else { const float2 t = inf[c]; xr = t.x; xi = t.y; }
    xr -= mur; xi -= mui;
    const float yr = wrr * xr + wri * xi, yi = wri * xr + wii * xi;
    const float4 wc = ((const float4*)w)[c];
    const float2 bb = bias[c];
    union { f16 h[2]; unsigned u; } p;
    p.h[0] = (f16)(wc.x * yr + wc.y * yi + bb.x);
    p.h[1] = (f16)(wc.z * yr + wc.w * yi + bb.y);
    *(unsigned*)(orow + 2 * c) = p.u;
  }
}

// Softmax over 512-length rows, in-place f16 (f32 math).
__global__ __launch_bounds__(256) void softmax512(f16* __restrict__ s) {
  const size_t row = blockIdx.x;
  f16* p = s + row * S_;
  const int tid = threadIdx.x;
  const float v0 = (float)p[tid], v1 = (float)p[tid + 256];
  __shared__ float red[256];
  red[tid] = fmaxf(v0, v1);
  __syncthreads();
  for (int t = 128; t > 0; t >>= 1) {
    if (tid < t) red[tid] = fmaxf(red[tid], red[tid + t]);
    __syncthreads();
  }
  const float mx = red[0];
  __syncthreads();
  const float e0 = __expf(v0 - mx), e1 = __expf(v1 - mx);
  red[tid] = e0 + e1;
  __syncthreads();
  for (int t = 128; t > 0; t >>= 1) {
    if (tid < t) red[tid] += red[tid + t];
    __syncthreads();
  }
  const float inv = 1.0f / red[0];
  p[tid] = (f16)(e0 * inv);
  p[tid + 256] = (f16)(e1 * inv);
}

// modrelu, in-place on f16 interleaved complex.
__global__ __launch_bounds__(256) void modrelu_k(f16* __restrict__ z,
                                                 const float* __restrict__ bmod,
                                                 long long n) {
  const long long i = (long long)blockIdx.x * 256 + threadIdx.x;
  if (i >= n) return;
  const float xr = (float)z[2 * i], xi = (float)z[2 * i + 1];
  const float mag = sqrtf(xr * xr + xi * xi);
  const float sc = fmaxf(mag + bmod[0], 0.0f) / (mag + 1e-12f);
  union { f16 h[2]; unsigned u; } p;
  p.h[0] = (f16)(sc * xr);
  p.h[1] = (f16)(sc * xi);
  *(unsigned*)(z + 2 * i) = p.u;
}

// ---------------------------------------------------------------------------
extern "C" void kernel_launch(void* const* d_in, const int* in_sizes, int n_in,
                              void* d_out, int out_size, void* d_ws,
                              size_t ws_size, hipStream_t stream) {
  (void)in_sizes; (void)n_in; (void)out_size; (void)ws_size;
  const float2* x     = (const float2*)d_in[0];
  const float*  ln1w  = (const float*)d_in[1];
  const float2* ln1b  = (const float2*)d_in[2];
  const float2* wq    = (const float2*)d_in[3];
  const float2* bq    = (const float2*)d_in[4];
  const float2* wk    = (const float2*)d_in[5];
  const float2* bk    = (const float2*)d_in[6];
  const float2* wv    = (const float2*)d_in[7];
  const float2* bv    = (const float2*)d_in[8];
  const float2* wo    = (const float2*)d_in[9];
  const float2* bo    = (const float2*)d_in[10];
  const float*  ln2w  = (const float*)d_in[11];
  const float2* ln2b  = (const float2*)d_in[12];
  const float2* w_in  = (const float2*)d_in[13];
  const float2* b_in  = (const float2*)d_in[14];
  const float*  ln3w  = (const float*)d_in[15];
  const float2* ln3b  = (const float2*)d_in[16];
  const float*  bmod  = (const float*)d_in[17];
  const float2* w_out = (const float2*)d_in[18];
  const float2* b_out = (const float2*)d_in[19];

  char* ws = (char*)d_ws;
  const size_t SZ16 = (size_t)ROWS_ * D2_ * sizeof(f16);  // 33.5 MB
  f16*    h16  = (f16*)(ws);
  f16*    q16  = (f16*)(ws + SZ16);
  f16*    k16  = (f16*)(ws + 2 * SZ16);
  f16*    v16b = (f16*)(ws + 3 * SZ16);
  f16*    ao16 = (f16*)(ws + 4 * SZ16);
  f16*    s16  = (f16*)(ws + 5 * SZ16);                       // 134.2 MB
  float2* xa   = (float2*)(ws + 5 * SZ16 + (size_t)B_ * H_ * S_ * S_ * 2);
  f16*    m16  = (f16*)((char*)xa + (size_t)ROWS_ * D_ * sizeof(float2));

  const dim3 blk(256);
  const dim3 gD(D_ / 64, ROWS_ / 128);   // (16, 64)
  const dim3 gM(M_ / 64, ROWS_ / 128);   // (64, 64)
  const dim3 gS(S_ / 64, S_ / 128, B_ * H_);   // (8, 4, 256)
  const dim3 gA(128 / 64, S_ / 128, B_ * H_);  // (2, 4, 256)

  // h = ln1(x)
  ln_cplx<false><<<ROWS_, blk, 0, stream>>>((const void*)x, ln1w, ln1b, h16, D_);
  // q,k,v = h@W + b  (f16 out)
  cgemm_wmma<true, false><<<gD, blk, 0, stream>>>(h16, wq, bq, nullptr, q16, D_, D_);
  cgemm_wmma<true, false><<<gD, blk, 0, stream>>>(h16, wk, bk, nullptr, k16, D_, D_);
  cgemm_wmma<true, false><<<gD, blk, 0, stream>>>(h16, wv, bv, nullptr, v16b, D_, D_);
  // scores + softmax + A*V
  qk_scores<<<gS, blk, 0, stream>>>(q16, k16, s16, 0.125f);
  softmax512<<<B_ * H_ * S_, blk, 0, stream>>>(s16);
  av_wmma<<<gA, blk, 0, stream>>>(s16, v16b, ao16);
  // x_attn = x + attn@wo + bo  (f32)
  cgemm_wmma<false, true><<<gD, blk, 0, stream>>>(ao16, wo, bo, x, (void*)xa, D_, D_);
  // h = ln2(x_attn); mid = h@w_in + b_in (f16)
  ln_cplx<false><<<ROWS_, blk, 0, stream>>>((const void*)xa, ln2w, ln2b, h16, D_);
  cgemm_wmma<true, false><<<gM, blk, 0, stream>>>(h16, w_in, b_in, nullptr, m16, D_, M_);
  // ln3 (in-place) + modrelu (in-place)
  ln_cplx<true><<<ROWS_, blk, 0, stream>>>((const void*)m16, ln3w, ln3b, m16, M_);
  const long long nmod = (long long)ROWS_ * M_;
  modrelu_k<<<(unsigned)((nmod + 255) / 256), blk, 0, stream>>>(m16, bmod, nmod);
  // out = x_attn + mid@w_out + b_out  (f32 -> d_out)
  cgemm_wmma<false, true><<<gD, blk, 0, stream>>>(m16, w_out, b_out, xa, d_out, M_, D_);
}
